// BeeNet_66262755443014
// MI455X (gfx1250) — matrix operand order repaired and checked by reading, not compiled, not run on hardware
//
#include <hip/hip_runtime.h>

// ---------------------------------------------------------------------------
// BeeNet (CommNet-style LSTM scan) for MI455X / gfx1250.
//   T=128, N=64 agents, HID=256, ACT=16, FEAT=3073.
// Kernel 1 (prep):  convert weights to bf16 ([n][k] layouts), zero state.
// Kernel 2 (eyes):  (8192x3073)x(3073x256) bf16-WMMA GEMM + bias + relu,
//                   B tile staged in LDS (double-buffered, shared by the 8
//                   waves of the WG); memory-bound: 100MB states @ 23.3 TB/s.
// Kernel 3 (scan):  persistent, 16 WGs x 8 waves, device-wide barriers,
//                   software-pipelined WMMA K-loops.
//                   comm einsum collapsed to comm[b,:] = -N*||cv_b||^2 *
//                   (sum_k mask[b,k]^2) * cv[b,:].
// ---------------------------------------------------------------------------

typedef __attribute__((ext_vector_type(16))) __bf16        v16bf;
typedef __attribute__((ext_vector_type(8)))  float         v8f;
typedef __attribute__((ext_vector_type(4)))  unsigned int  u32x4;

#define T_STEPS 128
#define NAG     64
#define HID     256
#define ACT     16
#define FEAT    3073
#define FEATP   3104      // FEAT padded to multiple of 32
#define KCAT    768       // [eye(256) | comm(256) | h(256)]
#define NGATE   1024
#define NWG     16        // workgroups in persistent scan kernel
#define NKCH    97        // FEATP/32 K-chunks in eyes GEMM

// ---- workspace layout (bytes, all 256-aligned) ----
#define OFF_BAR    0u
#define OFF_W1B    256u                                   // 256*3104*2
#define OFF_WCAT   (OFF_W1B   + 256u*FEATP*2u)            // 1024*768*2
#define OFF_WCB    (OFF_WCAT  + (unsigned)NGATE*KCAT*2u)  // 256*256*2
#define OFF_WAB    (OFF_WCB   + (unsigned)HID*HID*2u)     // 16*256*2
#define OFF_BCAT   (OFF_WAB   + (unsigned)ACT*HID*2u)     // 1024*4
#define OFF_EYES   (OFF_BCAT  + (unsigned)NGATE*4u)       // 8192*256*2
#define OFF_COMM   (OFF_EYES  + 8192u*HID*2u)             // 64*256*2
#define OFF_HBF    (OFF_COMM  + (unsigned)NAG*HID*2u)     // 64*256*2
#define OFF_C      (OFF_HBF   + (unsigned)NAG*HID*2u)     // 64*256*4
#define OFF_CV     (OFF_C     + (unsigned)NAG*HID*4u)     // 64*256*4
#define OFF_GATES  (OFF_CV    + (unsigned)NAG*HID*4u)     // 64*1024*4
#define OFF_AQ     (OFF_GATES + (unsigned)NAG*NGATE*4u)   // 64*16*4

// ---------------------------------------------------------------------------
__device__ inline unsigned short f2bf(float x) {
  union { float f; unsigned u; } v; v.f = x;
  unsigned r = v.u + 0x7FFFu + ((v.u >> 16) & 1u);   // round-to-nearest-even
  return (unsigned short)(r >> 16);
}
__device__ inline float bf2f(unsigned short h) {
  union { unsigned u; float f; } v; v.u = ((unsigned)h) << 16; return v.f;
}

union Frag { u32x4 u[2]; v16bf v; unsigned short s[16]; };

__device__ inline v8f vzero8() {
  v8f z;
  #pragma unroll
  for (int i = 0; i < 8; ++i) z[i] = 0.f;
  return z;
}

__device__ inline v8f wmma_bf(v16bf a, v16bf b, v8f c) {
  return __builtin_amdgcn_wmma_f32_16x16x32_bf16(
      /*neg_a=*/false, a, /*neg_b=*/false, b,
      /*c_mod=*/(short)0, c, /*reuse_a=*/false, /*reuse_b=*/false);
}

// A-matrix fragment (16x32 bf16, row-major source, row stride ld elements).
// Contiguous-K layout used here (same as B): lane L -> row (L&15); K in [koff,koff+16),
// koff = 16*(L>>4).  Two contiguous 16B loads per lane.
__device__ inline v16bf load_a_bf(const unsigned short* A, int ld, int m0,
                                  int kb, int lane) {
  int row = m0 + (lane & 15);
  int k   = kb + ((lane >> 4) << 4);
  const u32x4* p = (const u32x4*)(A + (size_t)row * ld + k);
  Frag f; f.u[0] = p[0]; f.u[1] = p[1];   // +8 bf16 == +1 u32x4
  return f.v;
}

// B-matrix fragment (32x16 bf16). Source stored [n][k] row-major (== the
// natural W[n_out][k_in] layout), so each lane's 16 K-values are contiguous:
// lane L -> col (L&15), K in [16*(L>>4), +16).
__device__ inline v16bf load_b_bf(const unsigned short* W, int ld, int n0,
                                  int kb, int lane) {
  int col = n0 + (lane & 15);
  int k   = kb + ((lane >> 4) << 4);
  const u32x4* p = (const u32x4*)(W + (size_t)col * ld + k);
  Frag f; f.u[0] = p[0]; f.u[1] = p[1];
  return f.v;
}

// A fragment from fp32 source with on-the-fly bf16 conversion (+K guard).
__device__ inline v16bf load_a_f32(const float* S, int ld, int m0, int kb,
                                   int lane, bool guard) {
  int row = m0 + (lane & 15);
  int k0  = kb + ((lane >> 4) << 4);
  const float* r = S + (size_t)row * ld;
  Frag f;
  #pragma unroll
  for (int j = 0; j < 8; ++j) {
    int ka = k0 + j, kc = k0 + 8 + j;
    float xa = (!guard || ka < FEAT) ? r[ka] : 0.f;
    float xc = (!guard || kc < FEAT) ? r[kc] : 0.f;
    f.s[j] = f2bf(xa); f.s[8 + j] = f2bf(xc);
  }
  return f.v;
}

__device__ inline float wsum(float x) {
  #pragma unroll
  for (int off = 16; off > 0; off >>= 1) x += __shfl_xor(x, off, 32);
  return x;
}
__device__ inline float sigm(float x)   { return 1.f / (1.f + __expf(-x)); }
__device__ inline float tanh_f(float x) { float e = __expf(-2.f * x);
                                          return (1.f - e) / (1.f + e); }

__device__ inline void grid_sync(unsigned* bar, unsigned target) {
  __threadfence();
  __syncthreads();
  if (threadIdx.x == 0) {
    __hip_atomic_fetch_add(bar, 1u, __ATOMIC_RELAXED, __HIP_MEMORY_SCOPE_AGENT);
    while (__hip_atomic_load(bar, __ATOMIC_RELAXED, __HIP_MEMORY_SCOPE_AGENT)
           < target)
      __builtin_amdgcn_s_sleep(1);
  }
  __syncthreads();
  __threadfence();
}

// ---------------------------------------------------------------------------
// Kernel 1: weight conversion / state init (idempotent; re-run every launch).
// ---------------------------------------------------------------------------
__global__ void prep_kernel(const float* __restrict__ W1,
                            const float* __restrict__ W_ih,
                            const float* __restrict__ W_hh,
                            const float* __restrict__ b_ih,
                            const float* __restrict__ b_hh,
                            const float* __restrict__ Wc,
                            const float* __restrict__ Wa,
                            const float* __restrict__ h0,
                            unsigned char* __restrict__ ws) {
  unsigned short* W1b   = (unsigned short*)(ws + OFF_W1B);
  unsigned short* Wcatb = (unsigned short*)(ws + OFF_WCAT);
  unsigned short* Wcb   = (unsigned short*)(ws + OFF_WCB);
  unsigned short* Wab   = (unsigned short*)(ws + OFF_WAB);
  float*          bcat  = (float*)(ws + OFF_BCAT);
  float*          c     = (float*)(ws + OFF_C);
  float*          cv    = (float*)(ws + OFF_CV);
  unsigned short* h_bf  = (unsigned short*)(ws + OFF_HBF);
  unsigned*       bar   = (unsigned*)(ws + OFF_BAR);

  size_t tid = (size_t)blockIdx.x * blockDim.x + threadIdx.x;
  size_t str = (size_t)gridDim.x * blockDim.x;

  for (size_t i = tid; i < (size_t)HID * FEATP; i += str) {
    size_t n = i / FEATP, k = i % FEATP;
    W1b[i] = (k < FEAT) ? f2bf(W1[n * FEAT + k]) : (unsigned short)0;
  }
  for (size_t i = tid; i < (size_t)NGATE * KCAT; i += str) {
    size_t g = i / KCAT, k = i % KCAT;
    float v = (k < 512) ? W_ih[g * 512 + k] : W_hh[g * HID + (k - 512)];
    Wcatb[i] = f2bf(v);
  }
  for (size_t i = tid; i < (size_t)HID * HID; i += str) Wcb[i] = f2bf(Wc[i]);
  for (size_t i = tid; i < (size_t)ACT * HID; i += str) Wab[i] = f2bf(Wa[i]);
  for (size_t i = tid; i < NGATE; i += str) bcat[i] = b_ih[i] + b_hh[i];
  for (size_t i = tid; i < (size_t)NAG * HID; i += str) {
    c[i] = 0.f; cv[i] = 0.f; h_bf[i] = f2bf(h0[i]);
  }
  if (tid == 0) *bar = 0u;
}

// ---------------------------------------------------------------------------
// Kernel 2: eyes = relu(states @ W1^T + b1)  -> bf16 (8192 x 256)
// One wave per 16-row strip; all 8 waves of a WG share the B tile (full W1
// K-chunk, 32x256 bf16 = 16KB) staged in LDS, double-buffered.  Next chunk's
// global loads are issued before the current chunk's 16 WMMAs.
// ---------------------------------------------------------------------------
__global__ __launch_bounds__(256) void eyes_kernel(
    const float* __restrict__ states, const unsigned short* __restrict__ W1b,
    const float* __restrict__ b1, unsigned short* __restrict__ eyes_bf) {
  __shared__ unsigned short Bs[2][32 * 256];         // 2 x 16KB

  int lane  = threadIdx.x & 31;
  int strip = blockIdx.x * 8 + (threadIdx.x >> 5);   // 0..511
  int m0    = strip * 16;

  v8f acc[16];
  #pragma unroll
  for (int i = 0; i < 16; ++i) acc[i] = vzero8();

  // cooperative B-chunk fetch: thread tid owns output column n = tid,
  // loads its 32 contiguous bf16 K-values (64B) of W1b.
  u32x4 rb[4];
  {
    const u32x4* p = (const u32x4*)(W1b + (size_t)threadIdx.x * FEATP);
    rb[0] = p[0]; rb[1] = p[1]; rb[2] = p[2]; rb[3] = p[3];
    u32x4* q = (u32x4*)&Bs[0][threadIdx.x * 32];
    q[0] = rb[0]; q[1] = rb[1]; q[2] = rb[2]; q[3] = rb[3];
  }
  __syncthreads();

  v16bf a = load_a_f32(states, FEAT, m0, 0, lane, false);

  for (int it = 0; it < NKCH; ++it) {
    int buf = it & 1;
    int kb  = it * 32;
    bool more = (it + 1 < NKCH);

    if (more) {                                      // prefetch next B chunk
      const u32x4* p =
          (const u32x4*)(W1b + (size_t)threadIdx.x * FEATP + kb + 32);
      rb[0] = p[0]; rb[1] = p[1]; rb[2] = p[2]; rb[3] = p[3];
    }
    v16bf a_cur = a;
    if (more)                                        // prefetch next A frag
      a = load_a_f32(states, FEAT, m0, kb + 32, lane, (kb + 64) > FEAT);

    #pragma unroll
    for (int ct = 0; ct < 16; ++ct) {
      const u32x4* p = (const u32x4*)
          &Bs[buf][(ct * 16 + (lane & 15)) * 32 + ((lane >> 4) << 4)];
      Frag f; f.u[0] = p[0]; f.u[1] = p[1];
      acc[ct] = wmma_bf(a_cur, f.v, acc[ct]);
    }

    if (more) {                                      // publish next chunk
      u32x4* q = (u32x4*)&Bs[buf ^ 1][threadIdx.x * 32];
      q[0] = rb[0]; q[1] = rb[1]; q[2] = rb[2]; q[3] = rb[3];
      __syncthreads();
    }
  }

  #pragma unroll
  for (int ct = 0; ct < 16; ++ct) {
    int n = ct * 16 + (lane & 15);
    float bias = b1[n];
    #pragma unroll
    for (int v = 0; v < 8; ++v) {
      int m = m0 + v + ((lane >> 4) << 3);
      float val = acc[ct][v] + bias;
      eyes_bf[(size_t)m * HID + n] = f2bf(fmaxf(val, 0.f));
    }
  }
}

// ---------------------------------------------------------------------------
// Kernel 3: persistent LSTM scan, 16 WGs x 256 threads (128 waves).
// ---------------------------------------------------------------------------
__global__ __launch_bounds__(256) void scan_kernel(
    const float* __restrict__ comm_mask, const float* __restrict__ ba,
    const float* __restrict__ Wv, const float* __restrict__ bv,
    unsigned char* __restrict__ ws, float* __restrict__ qout) {
  const unsigned short* Wcatb  = (const unsigned short*)(ws + OFF_WCAT);
  const unsigned short* Wcb    = (const unsigned short*)(ws + OFF_WCB);
  const unsigned short* Wab    = (const unsigned short*)(ws + OFF_WAB);
  const float*          bcat   = (const float*)(ws + OFF_BCAT);
  const unsigned short* eyes_bf= (const unsigned short*)(ws + OFF_EYES);
  unsigned short*       comm_bf= (unsigned short*)(ws + OFF_COMM);
  unsigned short*       h_bf   = (unsigned short*)(ws + OFF_HBF);
  float*                c      = (float*)(ws + OFF_C);
  float*                cv     = (float*)(ws + OFF_CV);
  float*                gates  = (float*)(ws + OFF_GATES);
  float*                Aq     = (float*)(ws + OFF_AQ);
  unsigned*             bar    = (unsigned*)(ws + OFF_BAR);

  int lane = threadIdx.x & 31;
  int wid  = blockIdx.x * 8 + (threadIdx.x >> 5);        // 0..127
  int gtid = blockIdx.x * 256 + threadIdx.x;             // 0..4095
  unsigned gen = 0;

  for (int t = 0; t < T_STEPS; ++t) {
    // --- Phase A: comm scalar + comm_bf; Q-head for step t-1 (waves 0..63) --
    if (wid < NAG) {
      int b = wid;
      const float* cvb = cv + b * HID;
      float p = 0.f;
      #pragma unroll
      for (int h = lane; h < HID; h += 32) { float x = cvb[h]; p += x * x; }
      p = wsum(p);                                        // ||cv_b||^2
      const float* mr = comm_mask + ((size_t)t * NAG + b) * NAG;
      float m2 = mr[lane] * mr[lane] + mr[lane + 32] * mr[lane + 32];
      m2 = wsum(m2);                                      // sum_k mask^2
      float s = -(float)NAG * p * m2;
      unsigned short* cb = comm_bf + b * HID;
      #pragma unroll
      for (int h = lane; h < HID; h += 32) cb[h] = f2bf(s * cvb[h]);

      if (t > 0) {                                        // q_{t-1}
        const unsigned short* hb = h_bf + b * HID;        // h_{t-1}
        float vb = 0.f;
        #pragma unroll
        for (int h = lane; h < HID; h += 32) vb += bf2f(hb[h]) * Wv[h];
        vb = wsum(vb) + bv[0];
        float aq = (lane < ACT) ? Aq[b * ACT + lane] : 0.f;
        float mean = wsum(aq) * (1.f / ACT);
        if (lane < ACT)
          qout[((size_t)(t - 1) * NAG + b) * ACT + lane] = vb + aq - mean;
      }
    }
    gen++; grid_sync(bar, gen * NWG);

    // --- Phase B: gates = [eye|comm|h] @ Wcat^T + bcat  (all 128 waves) -----
    // Software-pipelined: next-iteration fragments load while WMMAs execute.
    {
      int mt = wid >> 5, nt0 = wid & 31;                  // 2 N-tiles / wave
      v8f acc0 = vzero8(), acc1 = vzero8();
      const unsigned short* eyes_t = eyes_bf + (size_t)t * NAG * HID;

      v16bf a  = load_a_bf(eyes_t, HID, mt * 16, 0, lane);
      v16bf b0 = load_b_bf(Wcatb, KCAT, nt0 * 16, 0, lane);
      v16bf b1 = load_b_bf(Wcatb, KCAT, (nt0 + 32) * 16, 0, lane);
      for (int kb = 0; kb < KCAT; kb += 32) {
        v16bf an = a, b0n = b0, b1n = b1;
        int kn = kb + 32;
        if (kn < KCAT) {
          const unsigned short* Ab; int k;
          if      (kn < 256) { Ab = eyes_t;  k = kn;       }
          else if (kn < 512) { Ab = comm_bf; k = kn - 256; }
          else               { Ab = h_bf;    k = kn - 512; }
          an  = load_a_bf(Ab, HID, mt * 16, k, lane);
          b0n = load_b_bf(Wcatb, KCAT, nt0 * 16, kn, lane);
          b1n = load_b_bf(Wcatb, KCAT, (nt0 + 32) * 16, kn, lane);
        }
        acc0 = wmma_bf(a, b0, acc0);
        acc1 = wmma_bf(a, b1, acc1);
        a = an; b0 = b0n; b1 = b1n;
      }
      int mb = mt * 16 + ((lane >> 4) << 3);
      int nA = nt0 * 16 + (lane & 15);
      int nB = (nt0 + 32) * 16 + (lane & 15);
      float bA = bcat[nA], bB = bcat[nB];
      #pragma unroll
      for (int v = 0; v < 8; ++v) {
        gates[(size_t)(mb + v) * NGATE + nA] = acc0[v] + bA;
        gates[(size_t)(mb + v) * NGATE + nB] = acc1[v] + bB;
      }
    }
    gen++; grid_sync(bar, gen * NWG);

    // --- Phase C: elementwise LSTM cell --------------------------------------
    for (int idx = gtid; idx < NAG * HID; idx += NWG * 256) {
      int b = idx >> 8, j = idx & 255;
      const float* g = gates + (size_t)b * NGATE + j;
      float xi = sigm(g[0]);
      float xf = sigm(g[256]);
      float xg = tanh_f(g[512]);
      float xo = sigm(g[768]);
      float cn = xf * c[idx] + xi * xg;
      c[idx] = cn;
      h_bf[idx] = f2bf(xo * tanh_f(cn));
    }
    gen++; grid_sync(bar, gen * NWG);

    // --- Phase D: cv = h@Wc^T (waves 0..63); Aq = h@Wa^T + ba (64..67) -------
    if (wid < 64) {
      int mt = wid >> 4, nt = wid & 15;
      v8f acc = vzero8();
      v16bf a = load_a_bf(h_bf, HID, mt * 16, 0, lane);
      v16bf b = load_b_bf(Wcb, HID, nt * 16, 0, lane);
      for (int kb = 0; kb < HID; kb += 32) {
        v16bf an = a, bn = b;
        if (kb + 32 < HID) {
          an = load_a_bf(h_bf, HID, mt * 16, kb + 32, lane);
          bn = load_b_bf(Wcb, HID, nt * 16, kb + 32, lane);
        }
        acc = wmma_bf(a, b, acc);
        a = an; b = bn;
      }
      int mb = mt * 16 + ((lane >> 4) << 3);
      int n  = nt * 16 + (lane & 15);
      #pragma unroll
      for (int v = 0; v < 8; ++v) cv[(size_t)(mb + v) * HID + n] = acc[v];
    } else if (wid < 68) {
      int mt = wid - 64;
      v8f acc = vzero8();
      v16bf a = load_a_bf(h_bf, HID, mt * 16, 0, lane);
      v16bf b = load_b_bf(Wab, HID, 0, 0, lane);
      for (int kb = 0; kb < HID; kb += 32) {
        v16bf an = a, bn = b;
        if (kb + 32 < HID) {
          an = load_a_bf(h_bf, HID, mt * 16, kb + 32, lane);
          bn = load_b_bf(Wab, HID, 0, kb + 32, lane);
        }
        acc = wmma_bf(a, b, acc);
        a = an; b = bn;
      }
      int mb = mt * 16 + ((lane >> 4) << 3);
      int n  = lane & 15;
      float bias = ba[n];
      #pragma unroll
      for (int v = 0; v < 8; ++v) Aq[(mb + v) * ACT + n] = acc[v] + bias;
    }
    gen++; grid_sync(bar, gen * NWG);
  }

  // --- Epilogue: q for t = T-1 ----------------------------------------------
  if (wid < NAG) {
    int b = wid;
    const unsigned short* hb = h_bf + b * HID;
    float vb = 0.f;
    #pragma unroll
    for (int h = lane; h < HID; h += 32) vb += bf2f(hb[h]) * Wv[h];
    vb = wsum(vb) + bv[0];
    float aq = (lane < ACT) ? Aq[b * ACT + lane] : 0.f;
    float mean = wsum(aq) * (1.f / ACT);
    if (lane < ACT)
      qout[((size_t)(T_STEPS - 1) * NAG + b) * ACT + lane] = vb + aq - mean;
  }
}

// ---------------------------------------------------------------------------
extern "C" void kernel_launch(void* const* d_in, const int* in_sizes, int n_in,
                              void* d_out, int out_size, void* d_ws,
                              size_t ws_size, hipStream_t stream) {
  (void)in_sizes; (void)n_in; (void)out_size; (void)ws_size;
  const float* states    = (const float*)d_in[0];
  const float* comm_mask = (const float*)d_in[1];
  const float* W1        = (const float*)d_in[2];
  const float* b1        = (const float*)d_in[3];
  const float* W_ih      = (const float*)d_in[4];
  const float* W_hh      = (const float*)d_in[5];
  const float* b_ih      = (const float*)d_in[6];
  const float* b_hh      = (const float*)d_in[7];
  const float* Wc        = (const float*)d_in[8];
  const float* Wa        = (const float*)d_in[9];
  const float* ba        = (const float*)d_in[10];
  const float* Wv        = (const float*)d_in[11];
  const float* bv        = (const float*)d_in[12];
  const float* h0        = (const float*)d_in[13];
  unsigned char* ws = (unsigned char*)d_ws;
  float* q = (float*)d_out;

  prep_kernel<<<1024, 256, 0, stream>>>(W1, W_ih, W_hh, b_ih, b_hh, Wc, Wa,
                                        h0, ws);
  eyes_kernel<<<64, 256, 0, stream>>>(states,
                                      (const unsigned short*)(ws + OFF_W1B),
                                      b1, (unsigned short*)(ws + OFF_EYES));
  scan_kernel<<<NWG, 256, 0, stream>>>(comm_mask, ba, Wv, bv, ws, q);
}
